// LSTMStressPredictor_944892805172
// MI455X (gfx1250) — matrix operand order repaired and checked by reading, not compiled, not run on hardware
//
#include <hip/hip_runtime.h>

// ---------------------------------------------------------------------------
// Fused 2-layer LSTM (B=256, T=2048, I=6, H=64) + FC head (64->64 relu ->6)
// One workgroup owns a 16-row batch tile and runs the whole time recurrence
// for both layers + head.  Hidden/cell state never leaves the WGP.
// Matrix ops: V_WMMA_F32_16X16X32_F16 (f16 in, f32 accumulate).
// All weight B-fragments are held in VGPRs (loaded once); only the evolving
// h-state A-fragments are re-read from LDS each step (2x ds_load_b128 each).
// Activations are branchless: tanh(x) = 2*sigmoid(2x)-1, sigmoid via
// v_exp_f32 + v_rcp_f32 (no IEEE division, no exec-mask branches).
// ---------------------------------------------------------------------------

typedef _Float16 v16h __attribute__((ext_vector_type(16)));
typedef _Float16 v8h  __attribute__((ext_vector_type(8)));
typedef float    v8f  __attribute__((ext_vector_type(8)));

#define TT   2048
#define BB   256
#define II   6
#define HH   64
#define OO   6
#define HSTR 72     // f16 row stride for 16x64 state tiles (bank spread, 16B aligned)
#define GSTR 264    // f32 row stride for 16x256 gate staging

// Branchless gate activation:
//   sigmoid: scale=1, mul=1, add=0   ->  1/(1+exp(-x))
//   tanh:    scale=2, mul=2, add=-1  ->  2/(1+exp(-2x)) - 1
__device__ __forceinline__ float gate_act(float x, float scale, float mul, float add) {
    const float e = __expf(-scale * x);                 // v_mul + v_exp_f32
    const float r = __builtin_amdgcn_rcpf(1.0f + e);    // v_add + v_rcp_f32
    return __builtin_fmaf(r, mul, add);                 // v_fma
}
__device__ __forceinline__ float tanh_fast(float x) {
    const float e = __expf(-2.0f * x);
    const float r = __builtin_amdgcn_rcpf(1.0f + e);
    return __builtin_fmaf(r, 2.0f, -1.0f);
}

// A-fragment (16x32 f16) for wmma_f32_16x16x32_f16 from an LDS [16][HSTR] tile.
// lane = 16*g + l holds row l; halves 0..7 -> k = kt*32 + 8g + h,
// halves 8..15 -> k = kt*32 + 16 + 8g + (h-8).  Two contiguous 16B runs.
__device__ __forceinline__ v16h load_a_frag(const _Float16* hb, int l, int g, int kt) {
    const int base = l * HSTR + kt * 32;
    v8h lo = *(const v8h*)(hb + base + g * 8);
    v8h hi = *(const v8h*)(hb + base + 16 + g * 8);
    v16h a;
#pragma unroll
    for (int j = 0; j < 8; ++j) { a[j] = lo[j]; a[8 + j] = hi[j]; }
    return a;
}

// B-fragment (32x16 f16): lane 16*g+l holds column n, h<8 -> k = kt*32 + 8g + h, h>=8 -> k = kt*32 + 16 + 8g + (h-8).
// W is row-major [N][K] (i.e. gates x hidden), so B[k][n] = W[n][k].
__device__ __forceinline__ v16h load_b_frag(const float* W, int Kdim, int n, int kt, int g) {
    const float* p = W + (size_t)n * Kdim + kt * 32 + g * 8;
    v16h b;
#pragma unroll
    for (int h = 0; h < 16; ++h) b[h] = (_Float16)p[h < 8 ? h : h + 8];
    return b;
}

__device__ __forceinline__ v8f wmma_step(v16h a, v16h b, v8f c) {
    return __builtin_amdgcn_wmma_f32_16x16x32_f16(false, a, false, b, (short)0, c,
                                                  false, false);
}

__global__ __launch_bounds__(256)
void lstm_fused_kernel(const float* __restrict__ x,
                       const float* __restrict__ Wih0, const float* __restrict__ Whh0,
                       const float* __restrict__ b0,
                       const float* __restrict__ Wih1, const float* __restrict__ Whh1,
                       const float* __restrict__ b1,
                       const float* __restrict__ W1,  const float* __restrict__ bf1,
                       const float* __restrict__ W2,  const float* __restrict__ bf2,
                       float* __restrict__ out) {
    __shared__ __align__(16) _Float16 h0buf[16 * HSTR];   // layer0 h state (f16)
    __shared__ __align__(16) _Float16 h1buf[16 * HSTR];   // layer1 h state (f16)
    __shared__ __align__(16) _Float16 fcbuf[16 * HSTR];   // relu(fc1) (f16)
    __shared__ __align__(16) float    gates[16 * GSTR];   // activated gate staging
    __shared__ __align__(16) float    xbuf[16 * 8];       // x_t tile (16x6, pad 8)
    __shared__ __align__(16) float    w2s[OO * HH];       // FC2 weights

    const int tid  = threadIdx.x;
    const int wv   = tid >> 5;          // wave 0..7
    const int lane = tid & 31;
    const int g    = lane >> 4;         // half-wave
    const int l    = lane & 15;
    const int m0   = blockIdx.x * 16;   // batch tile base

    // ---- one-time init --------------------------------------------------
    for (int i = tid; i < 16 * HSTR; i += 256) {
        h0buf[i] = (_Float16)0.0f;
        h1buf[i] = (_Float16)0.0f;
    }
    for (int i = tid; i < OO * HH; i += 256) w2s[i] = W2[i];

    // Persistent weight fragments in registers. Wave wv owns gate columns
    // [wv*32, wv*32+32) as two 16-col WMMA tiles tt = 2*wv, 2*wv+1.
    v16h Bh0[2][2], Bi1[2][2], Bh1[2][2];   // [n-tile][k-tile]
    float b0c[2], b1c[2], wih0r[2][II];
    float actS[2], actM[2], actA[2];         // branchless activation constants
#pragma unroll
    for (int j = 0; j < 2; ++j) {
        const int tt = 2 * wv + j;
        const int n  = tt * 16 + l;          // gate column 0..255
        b0c[j] = b0[n];
        b1c[j] = b1[n];
        const bool is_g = ((tt >> 2) == 2);  // tiles 8..11 = candidate gate (tanh)
        actS[j] = is_g ? 2.0f : 1.0f;
        actM[j] = is_g ? 2.0f : 1.0f;
        actA[j] = is_g ? -1.0f : 0.0f;
#pragma unroll
        for (int i = 0; i < II; ++i) wih0r[j][i] = Wih0[n * II + i];
#pragma unroll
        for (int kt = 0; kt < 2; ++kt) {
            Bh0[j][kt] = load_b_frag(Whh0, HH, n, kt, g);
            Bi1[j][kt] = load_b_frag(Wih1, HH, n, kt, g);
            Bh1[j][kt] = load_b_frag(Whh1, HH, n, kt, g);
        }
    }
    // FC1 fragments (used by waves 0..3; all waves load in-range rows).
    const int nfc = (wv & 3) * 16 + l;           // 0..63, valid row of W1
    const float bf1c = bf1[nfc];
    v16h Bw1_0 = load_b_frag(W1, HH, nfc, 0, g);
    v16h Bw1_1 = load_b_frag(W1, HH, nfc, 1, g);

    // Elementwise update ownership: thread -> (row um, cols un..un+3).
    const int um = tid & 15;
    const int un = (tid >> 4) * 4;
    float c0r[4] = {0.f, 0.f, 0.f, 0.f};
    float c1r[4] = {0.f, 0.f, 0.f, 0.f};

    // FC2 ownership (threads 0..95)
    const int fm = tid / OO;
    const int fo = tid - fm * OO;
    const float bf2v = (tid < 16 * OO) ? bf2[fo] : 0.0f;

    __syncthreads();

    // ---- time recurrence -------------------------------------------------
    for (int t = 0; t < TT; ++t) {
        // stage x_t tile (16 x 6)
        if (tid < 16 * II) {
            const int m = tid / II, i = tid - m * II;
            xbuf[m * 8 + i] = x[((size_t)(m0 + m) * TT + t) * II + i];
        }
        if (t + 1 < TT && tid < 16)
            __builtin_prefetch(&x[((size_t)(m0 + tid) * TT + (t + 1)) * II], 0, 1);
        __syncthreads();

        // ---- layer 0: gates = x_t@Wih0^T + b0 + h0_{t-1}@Whh0^T ----------
        {
            const v16h a0 = load_a_frag(h0buf, l, g, 0);
            const v16h a1 = load_a_frag(h0buf, l, g, 1);
#pragma unroll
            for (int j = 0; j < 2; ++j) {
                v8f acc;
#pragma unroll
                for (int r = 0; r < 8; ++r) {
                    const int row = r + 8 * g;
                    float xin = b0c[j];
#pragma unroll
                    for (int i = 0; i < II; ++i) xin += xbuf[row * 8 + i] * wih0r[j][i];
                    acc[r] = xin;
                }
                acc = wmma_step(a0, Bh0[j][0], acc);
                acc = wmma_step(a1, Bh0[j][1], acc);
                const int col = (2 * wv + j) * 16 + l;
#pragma unroll
                for (int r = 0; r < 8; ++r)
                    gates[(r + 8 * g) * GSTR + col] =
                        gate_act(acc[r], actS[j], actM[j], actA[j]);
            }
        }
        __syncthreads();
        // h0/c0 elementwise update
#pragma unroll
        for (int q = 0; q < 4; ++q) {
            const int n = un + q;
            const float iv = gates[um * GSTR + n];
            const float fv = gates[um * GSTR + n + 64];
            const float gv = gates[um * GSTR + n + 128];
            const float ov = gates[um * GSTR + n + 192];
            c0r[q] = __builtin_fmaf(fv, c0r[q], iv * gv);
            h0buf[um * HSTR + n] = (_Float16)(ov * tanh_fast(c0r[q]));
        }
        __syncthreads();

        // ---- layer 1: gates = h0_t@Wih1^T + b1 + h1_{t-1}@Whh1^T ---------
        {
            const v16h a0 = load_a_frag(h0buf, l, g, 0);
            const v16h a1 = load_a_frag(h0buf, l, g, 1);
            const v16h p0 = load_a_frag(h1buf, l, g, 0);
            const v16h p1 = load_a_frag(h1buf, l, g, 1);
#pragma unroll
            for (int j = 0; j < 2; ++j) {
                v8f acc;
#pragma unroll
                for (int r = 0; r < 8; ++r) acc[r] = b1c[j];
                acc = wmma_step(a0, Bi1[j][0], acc);
                acc = wmma_step(a1, Bi1[j][1], acc);
                acc = wmma_step(p0, Bh1[j][0], acc);
                acc = wmma_step(p1, Bh1[j][1], acc);
                const int col = (2 * wv + j) * 16 + l;
#pragma unroll
                for (int r = 0; r < 8; ++r)
                    gates[(r + 8 * g) * GSTR + col] =
                        gate_act(acc[r], actS[j], actM[j], actA[j]);
            }
        }
        __syncthreads();
        // h1/c1 elementwise update
#pragma unroll
        for (int q = 0; q < 4; ++q) {
            const int n = un + q;
            const float iv = gates[um * GSTR + n];
            const float fv = gates[um * GSTR + n + 64];
            const float gv = gates[um * GSTR + n + 128];
            const float ov = gates[um * GSTR + n + 192];
            c1r[q] = __builtin_fmaf(fv, c1r[q], iv * gv);
            h1buf[um * HSTR + n] = (_Float16)(ov * tanh_fast(c1r[q]));
        }
        __syncthreads();

        // ---- FC1: relu(h1_t @ W1^T + bf1), waves 0..3 own 16 cols each ---
        if (wv < 4) {
            const v16h a0 = load_a_frag(h1buf, l, g, 0);
            const v16h a1 = load_a_frag(h1buf, l, g, 1);
            v8f acc;
#pragma unroll
            for (int r = 0; r < 8; ++r) acc[r] = bf1c;
            acc = wmma_step(a0, Bw1_0, acc);
            acc = wmma_step(a1, Bw1_1, acc);
            const int col = wv * 16 + l;
#pragma unroll
            for (int r = 0; r < 8; ++r)
                fcbuf[(r + 8 * g) * HSTR + col] = (_Float16)fmaxf(acc[r], 0.0f);
        }
        __syncthreads();

        // ---- FC2: (16x64)@(64x6) + bf2, scalar, threads 0..95 ------------
        if (tid < 16 * OO) {
            float s = bf2v;
#pragma unroll
            for (int k = 0; k < HH; ++k)
                s += (float)fcbuf[fm * HSTR + k] * w2s[fo * HH + k];
            out[((size_t)(m0 + fm) * TT + t) * OO + fo] = s;
        }
        // no trailing barrier needed: next write to fcbuf/gates is ordered
        // behind multiple barriers in the next iteration.
    }
}

extern "C" void kernel_launch(void* const* d_in, const int* in_sizes, int n_in,
                              void* d_out, int out_size, void* d_ws, size_t ws_size,
                              hipStream_t stream) {
    (void)in_sizes; (void)n_in; (void)out_size; (void)d_ws; (void)ws_size;
    const float* x    = (const float*)d_in[0];
    const float* Wih0 = (const float*)d_in[1];
    const float* Whh0 = (const float*)d_in[2];
    const float* b0   = (const float*)d_in[3];
    const float* Wih1 = (const float*)d_in[4];
    const float* Whh1 = (const float*)d_in[5];
    const float* b1   = (const float*)d_in[6];
    const float* W1   = (const float*)d_in[7];
    const float* bf1  = (const float*)d_in[8];
    const float* W2   = (const float*)d_in[9];
    const float* bf2  = (const float*)d_in[10];
    float* out = (float*)d_out;

    dim3 grid(BB / 16);   // 16 workgroups, one per 16-row batch tile
    dim3 block(256);      // 8 wave32
    hipLaunchKernelGGL(lstm_fused_kernel, grid, block, 0, stream,
                       x, Wih0, Whh0, b0, Wih1, Whh1, b1, W1, bf1, W2, bf2, out);
}
